// NoCeptionNet_28097676051181
// MI455X (gfx1250) — compile-verified, matrix operand order checked by reading, not run on hardware
//
#include <hip/hip_runtime.h>
#include <hip/hip_bf16.h>
#include <math.h>

typedef __attribute__((ext_vector_type(16))) _Float16 v16h;
typedef __attribute__((ext_vector_type(8)))  _Float16 v8h;
typedef __attribute__((ext_vector_type(8)))  float    v8f;
typedef __attribute__((ext_vector_type(4)))  float    v4f;

#define HD 64
#define H2 2048
#define WFRAG (128 * 32 * 32)   // halves per shuffled weight array (= 64*2048)

// ---------------------------------------------------------------------------
// Monotonic IEEE float <-> uint bijection (branchless): order-preserving.
// Segment-max = ONE unconditional global_atomic_max_u32.
// fmono(-inf) = 0x007FFFFF is the "untouched" sentinel.
// ---------------------------------------------------------------------------
#define MONO_NEG_INF 0x007FFFFFu
__device__ __forceinline__ unsigned int fmono(float v) {
    unsigned int u = __float_as_uint(v);
    unsigned int s = (unsigned int)(((int)u) >> 31);
    return u ^ (s | 0x80000000u);
}
__device__ __forceinline__ float fmono_inv(unsigned int u) {
    unsigned int s = (unsigned int)(((int)u) >> 31);
    return __uint_as_float(u ^ ((~s) | 0x80000000u));
}

__device__ __forceinline__ float elu1(float x) {
    return x > 0.0f ? x : (expf(x) - 1.0f);
}

// ---------------------------------------------------------------------------
// Input projection: out[r, o] = elu(in[r,:13] . w[:,o] + b[o])
// ---------------------------------------------------------------------------
__global__ __launch_bounds__(256) void proj_elu_kernel(
    const float* __restrict__ in, const float* __restrict__ w,
    const float* __restrict__ b, float* __restrict__ out, int rows)
{
    int idx = blockIdx.x * 256 + threadIdx.x;
    if (idx >= rows * HD) return;
    int r = idx >> 6, o = idx & 63;
    const float* ir = in + r * 13;
    float acc = b[o];
#pragma unroll
    for (int k = 0; k < 13; ++k) acc = fmaf(ir[k], w[k * HD + o], acc);
    out[idx] = elu1(acc);
}

// ---------------------------------------------------------------------------
// LayerNorm over rows of 64; one wave per row, 2 elems per lane.
// ---------------------------------------------------------------------------
__global__ __launch_bounds__(256) void ln_rows_kernel(
    const float* __restrict__ x, const float* __restrict__ g,
    const float* __restrict__ b, float* __restrict__ o32,
    _Float16* __restrict__ o16, int rows)
{
    int row = blockIdx.x * 8 + (threadIdx.x >> 5);
    if (row >= rows) return;
    int lane = threadIdx.x & 31;
    const float* xr = x + (size_t)row * HD;
    float v0 = xr[lane], v1 = xr[lane + 32];
    float s = v0 + v1;
    for (int m = 16; m; m >>= 1) s += __shfl_xor(s, m, 32);
    float mean = s * (1.0f / 64.0f);
    float d0 = v0 - mean, d1 = v1 - mean;
    float q = d0 * d0 + d1 * d1;
    for (int m = 16; m; m >>= 1) q += __shfl_xor(q, m, 32);
    float inv = rsqrtf(q * (1.0f / 64.0f) + 1e-5f);
    float y0 = d0 * inv * g[lane] + b[lane];
    float y1 = d1 * inv * g[lane + 32] + b[lane + 32];
    if (o32) { o32[(size_t)row * HD + lane] = y0; o32[(size_t)row * HD + lane + 32] = y1; }
    if (o16) { o16[(size_t)row * HD + lane] = (_Float16)y0; o16[(size_t)row * HD + lane + 32] = (_Float16)y1; }
}

// ---------------------------------------------------------------------------
// Pre-shuffle one (64 x 2048) f32 weight array into WMMA-B fragment order:
// out[((ct*32 + lane)*32) + i] = b-frag half i for column tile ct, lane.
// ---------------------------------------------------------------------------
__global__ __launch_bounds__(256) void wshuffle_kernel(
    const float* __restrict__ W, _Float16* __restrict__ out)
{
    int idx = blockIdx.x * 256 + threadIdx.x;
    if (idx >= WFRAG) return;
    int i    = idx & 31;
    int lane = (idx >> 5) & 31;
    int ct   = idx >> 10;
    int j     = ct * 16 + (lane & 15);
    int khalf = (lane >> 4) * 8;
    int kb    = i >> 4, ii = i & 15;
    int K = kb * 32 + khalf + (ii < 8 ? ii : 16 + (ii - 8));
    out[idx] = (_Float16)W[K * H2 + j];
}

// ---------------------------------------------------------------------------
// Vectorized fill (uint4) — init agg to fmono(-inf)
// ---------------------------------------------------------------------------
__global__ __launch_bounds__(256) void fill_u32x4_kernel(
    uint4* __restrict__ p, unsigned int v, int n4)
{
    int i = blockIdx.x * 256 + threadIdx.x;
    if (i < n4) p[i] = make_uint4(v, v, v, v);
}

// ---------------------------------------------------------------------------
// Fused NNConv core (one direction, one layer):
//   We_tile(16x2048) = ef_tile(16x64) @ W(64x2048) + bias   [WMMA f16->f32,
//     bias in the C operand]
//   msg = nf[gather[e], h] * We[e, j];  atomic segment-max into agg (mono u32)
// Tail edges scatter to dummy row N (no branches).  agg is L2-resident.
// ---------------------------------------------------------------------------
__global__ __launch_bounds__(128) void nnconv_fused_kernel(
    const _Float16* __restrict__ ef16,   // E x 64 (f16)
    const _Float16* __restrict__ w16s,   // shuffled frags, WFRAG halves
    const float*    __restrict__ bias,   // 2048
    const float*    __restrict__ nf,     // N x 64 (f32)
    const int*      __restrict__ gidx,   // E (gather)
    const int*      __restrict__ sidx,   // E (scatter)
    unsigned int*   __restrict__ agg,    // (N+1) x 2048, pre-filled sentinel
    int E, int N)
{
    __shared__ __align__(16) float s_nf_t[16 * HD];  // [h][m] transposed
    __shared__ int s_sc[16];

    const int e0  = blockIdx.x * 16;
    const int tid = threadIdx.x;

    if (tid < 16) {
        int e = e0 + tid;
        s_sc[tid] = (e < E) ? sidx[e] : N;   // tail -> dummy sink row
    }
    for (int i = tid; i < 16 * HD; i += 128) {
        int h = i >> 4, m = i & 15;
        int e = e0 + m; if (e >= E) e = E - 1;
        s_nf_t[i] = nf[(size_t)gidx[e] * HD + h];
    }
    __syncthreads();

    const int lane  = tid & 31;
    const int wave  = tid >> 5;
    const int lrow  = lane & 15;
    const int khalf = (lane >> 4) * 8;
    const int mrow  = (lane >> 4) * 8;

    // Per-lane scatter offsets for the 8 rows this lane owns (regs, no LDS
    // in the hot loop; 32-bit offsets -> saddr+voffset atomics).
    unsigned int aoff[8];
#pragma unroll
    for (int v = 0; v < 8; ++v) aoff[v] = (unsigned int)s_sc[mrow + v] * H2;

    // A fragments: 4 aligned 16-byte loads from the ef row.
    v16h a0, a1;
    {
        int er = e0 + lrow; if (er >= E) er = E - 1;
        const v8h* rp = (const v8h*)(ef16 + (size_t)er * HD);
        int gb = khalf >> 3;
        v8h lo0 = rp[gb],     hi0 = rp[gb + 2];
        v8h lo1 = rp[gb + 4], hi1 = rp[gb + 6];
        a0 = __builtin_shufflevector(lo0, hi0, 0,1,2,3,4,5,6,7,8,9,10,11,12,13,14,15);
        a1 = __builtin_shufflevector(lo1, hi1, 0,1,2,3,4,5,6,7,8,9,10,11,12,13,14,15);
    }

    for (int ct = wave; ct < 128; ct += 4) {
        const int j = ct * 16 + lrow;
        const v16h* bp = (const v16h*)(w16s + ((size_t)(ct * 32 + lane)) * 32);
        v16h b0 = bp[0];
        v16h b1 = bp[1];

        const float bj = bias[j];
        v8f c = { bj, bj, bj, bj, bj, bj, bj, bj };
        c = __builtin_amdgcn_wmma_f32_16x16x32_f16(false, a0, false, b0,
                                                   (short)0, c, false, false);
        c = __builtin_amdgcn_wmma_f32_16x16x32_f16(false, a1, false, b1,
                                                   (short)0, c, false, false);

        const int h = ct >> 1;            // 16 columns share one h (j/32)
        // 8 gathered nf values for this lane's rows: two ds_load_b128
        const v4f nf0 = *(const v4f*)&s_nf_t[h * 16 + mrow];
        const v4f nf1 = *(const v4f*)&s_nf_t[h * 16 + mrow + 4];
#pragma unroll
        for (int v = 0; v < 4; ++v)
            atomicMax(agg + aoff[v] + j, fmono(c[v] * nf0[v]));
#pragma unroll
        for (int v = 0; v < 4; ++v)
            atomicMax(agg + aoff[v + 4] + j, fmono(c[v + 4] * nf1[v]));
    }
}

// ---------------------------------------------------------------------------
// Reduce: out[n,c] = sum_h decode(agg[n,h,c]) + cb[c];  sentinel -> 0
// ---------------------------------------------------------------------------
__global__ __launch_bounds__(256) void nnconv_reduce_kernel(
    const unsigned int* __restrict__ agg, const float* __restrict__ cb,
    float* __restrict__ out, int N)
{
    int idx = blockIdx.x * 256 + threadIdx.x;
    if (idx >= N * 32) return;
    int n = idx >> 5, c = idx & 31;
    const unsigned int* p = agg + (size_t)n * H2 + c;
    float s = 0.0f;
#pragma unroll
    for (int h = 0; h < HD; ++h) {
        unsigned int u = p[h * 32];
        s += (u == MONO_NEG_INF) ? 0.0f : fmono_inv(u);
    }
    out[idx] = s + cb[c];
}

// ---------------------------------------------------------------------------
// m = LN(concat(mi, mo)); h_n = elu(nf + m).  One node per 64-thread block.
// ---------------------------------------------------------------------------
__global__ __launch_bounds__(64) void combine_ln_elu_kernel(
    const float* __restrict__ nf, const float* __restrict__ mi,
    const float* __restrict__ mo, const float* __restrict__ mg,
    const float* __restrict__ mb, float* __restrict__ hn, int N)
{
    int n = blockIdx.x;
    if (n >= N) return;
    int t = threadIdx.x;
    float v = (t < 32) ? mi[n * 32 + t] : mo[n * 32 + (t - 32)];
    __shared__ float red[64];
    red[t] = v;
    __syncthreads();
    float s = 0.0f;
#pragma unroll
    for (int k = 0; k < 64; ++k) s += red[k];
    float mean = s * (1.0f / 64.0f);
    float d = v - mean;
    __syncthreads();
    red[t] = d * d;
    __syncthreads();
    float q = 0.0f;
#pragma unroll
    for (int k = 0; k < 64; ++k) q += red[k];
    float inv = rsqrtf(q * (1.0f / 64.0f) + 1e-5f);
    float y = d * inv * mg[t] + mb[t];
    float x = nf[(size_t)n * HD + t] + y;
    hn[(size_t)n * HD + t] = elu1(x);
}

// ---------------------------------------------------------------------------
// Column-wise max pool: g[k] = max_n h_n[n,k]
// ---------------------------------------------------------------------------
__global__ __launch_bounds__(256) void colmax_kernel(
    const float* __restrict__ hn, float* __restrict__ g, int N)
{
    int k = blockIdx.x;
    int t = threadIdx.x;
    float m = -__builtin_inff();
    for (int n = t; n < N; n += 256) m = fmaxf(m, hn[(size_t)n * HD + k]);
    __shared__ float red[256];
    red[t] = m;
    __syncthreads();
    for (int s = 128; s; s >>= 1) {
        if (t < s) red[t] = fmaxf(red[t], red[t + s]);
        __syncthreads();
    }
    if (t == 0) g[k] = red[0];
}

// ---------------------------------------------------------------------------
// Head: out = elu(g @ f1_w + f1_b) @ f2_w + f2_b
// ---------------------------------------------------------------------------
__global__ __launch_bounds__(256) void head_kernel(
    const float* __restrict__ g, const float* __restrict__ f1w,
    const float* __restrict__ f1b, const float* __restrict__ f2w,
    const float* __restrict__ f2b, float* __restrict__ out)
{
    __shared__ float sg[64];
    __shared__ float red[256];
    int t = threadIdx.x;
    if (t < 64) sg[t] = g[t];
    __syncthreads();
    float acc = f1b[t];
#pragma unroll
    for (int k = 0; k < 64; ++k) acc = fmaf(sg[k], f1w[k * 256 + t], acc);
    red[t] = elu1(acc) * f2w[t];
    __syncthreads();
    for (int s = 128; s; s >>= 1) {
        if (t < s) red[t] += red[t + s];
        __syncthreads();
    }
    if (t == 0) out[0] = red[0] + f2b[0];
}

// ---------------------------------------------------------------------------
extern "C" void kernel_launch(void* const* d_in, const int* in_sizes, int n_in,
                              void* d_out, int out_size, void* d_ws, size_t ws_size,
                              hipStream_t stream)
{
    const float* node_inp = (const float*)d_in[0];
    const float* edge_inp = (const float*)d_in[1];
    const int*   src      = (const int*)d_in[2];
    const int*   dst      = (const int*)d_in[3];
    const float* node_w   = (const float*)d_in[4];
    const float* node_b   = (const float*)d_in[5];
    const float* edge_w   = (const float*)d_in[6];
    const float* edge_b   = (const float*)d_in[7];
    const float* mi_w     = (const float*)d_in[8];    // (2, 64, 2048)
    const float* mi_b     = (const float*)d_in[9];    // (2, 2048)
    const float* mo_w     = (const float*)d_in[10];   // (2, 64, 2048)
    const float* mo_b     = (const float*)d_in[11];   // (2, 2048)
    const float* mi_cb    = (const float*)d_in[12];   // (2, 32)
    const float* mo_cb    = (const float*)d_in[13];   // (2, 32)
    const float* nn_g     = (const float*)d_in[14];   // (2, 64)
    const float* nn_b     = (const float*)d_in[15];
    const float* mn_g     = (const float*)d_in[16];
    const float* mn_b     = (const float*)d_in[17];
    const float* f1_w     = (const float*)d_in[18];   // (64, 256)
    const float* f1_b     = (const float*)d_in[19];
    const float* f2_w     = (const float*)d_in[20];   // (256, 1)
    const float* f2_b     = (const float*)d_in[21];

    const int N = in_sizes[0] / 13;
    const int E = in_sizes[1] / 13;

    // ---- workspace carve-up (256B aligned) ----
    char* ws = (char*)d_ws;
    size_t off = 0;
    auto carve = [&](size_t bytes) {
        void* p = ws + off;
        off += (bytes + 255) & ~(size_t)255;
        return p;
    };
    float*        h_n    = (float*)       carve((size_t)N * HD * 4);
    float*        h_e    = (float*)       carve((size_t)E * HD * 4);
    float*        nf     = (float*)       carve((size_t)N * HD * 4);
    _Float16*     ef16   = (_Float16*)    carve((size_t)E * HD * 2);
    _Float16*     w16s   = (_Float16*)    carve((size_t)4 * WFRAG * 2);  // mi0,mi1,mo0,mo1
    unsigned int* agg    = (unsigned int*)carve((size_t)(N + 1) * H2 * 4); // +1 sink row
    float*        mi_buf = (float*)       carve((size_t)N * 32 * 4);
    float*        mo_buf = (float*)       carve((size_t)N * 32 * 4);
    float*        gbuf   = (float*)       carve(64 * 4);

    // ---- input projections + ELU ----
    proj_elu_kernel<<<(N * HD + 255) / 256, 256, 0, stream>>>(node_inp, node_w, node_b, h_n, N);
    proj_elu_kernel<<<(E * HD + 255) / 256, 256, 0, stream>>>(edge_inp, edge_w, edge_b, h_e, E);

    // ---- weight pre-shuffle into WMMA-B fragment layout (f16) ----
    const int shufGrid = (WFRAG + 255) / 256;
    wshuffle_kernel<<<shufGrid, 256, 0, stream>>>(mi_w,           w16s);
    wshuffle_kernel<<<shufGrid, 256, 0, stream>>>(mi_w + HD * H2, w16s + WFRAG);
    wshuffle_kernel<<<shufGrid, 256, 0, stream>>>(mo_w,           w16s + 2 * WFRAG);
    wshuffle_kernel<<<shufGrid, 256, 0, stream>>>(mo_w + HD * H2, w16s + 3 * WFRAG);

    const int aggN4     = ((N + 1) * H2) / 4;
    const int edgeTiles = (E + 15) / 16;

    for (int l = 0; l < 2; ++l) {
        ln_rows_kernel<<<(N + 7) / 8, 256, 0, stream>>>(h_n, nn_g + l * HD, nn_b + l * HD,
                                                        nf, (_Float16*)nullptr, N);
        ln_rows_kernel<<<(E + 7) / 8, 256, 0, stream>>>(h_e, nn_g + l * HD, nn_b + l * HD,
                                                        (float*)nullptr, ef16, E);
        // mi: gather=src, scatter=dst
        fill_u32x4_kernel<<<(aggN4 + 255) / 256, 256, 0, stream>>>((uint4*)agg, MONO_NEG_INF, aggN4);
        nnconv_fused_kernel<<<edgeTiles, 128, 0, stream>>>(ef16, w16s + (size_t)l * WFRAG,
                                                           mi_b + l * H2, nf, src, dst, agg, E, N);
        nnconv_reduce_kernel<<<(N * 32 + 255) / 256, 256, 0, stream>>>(agg, mi_cb + l * 32, mi_buf, N);
        // mo: gather=dst, scatter=src
        fill_u32x4_kernel<<<(aggN4 + 255) / 256, 256, 0, stream>>>((uint4*)agg, MONO_NEG_INF, aggN4);
        nnconv_fused_kernel<<<edgeTiles, 128, 0, stream>>>(ef16, w16s + (size_t)(2 + l) * WFRAG,
                                                           mo_b + l * H2, nf, dst, src, agg, E, N);
        nnconv_reduce_kernel<<<(N * 32 + 255) / 256, 256, 0, stream>>>(agg, mo_cb + l * 32, mo_buf, N);
        // concat -> LN -> residual -> ELU
        combine_ln_elu_kernel<<<N, 64, 0, stream>>>(nf, mi_buf, mo_buf,
                                                    mn_g + l * HD, mn_b + l * HD, h_n, N);
    }

    colmax_kernel<<<64, 256, 0, stream>>>(h_n, gbuf, N);
    head_kernel<<<1, 256, 0, stream>>>(gbuf, f1_w, f1_b, f2_w, f2_b, (float*)d_out);
}
